// LocalAttentionGate_50362786512970
// MI455X (gfx1250) — matrix operand order repaired and checked by reading, not compiled, not run on hardware
//
#include <hip/hip_runtime.h>
#include <stdint.h>

typedef __bf16 bf16_t;
typedef __attribute__((ext_vector_type(16))) __bf16 v16bf;
typedef __attribute__((ext_vector_type(8))) float v8f;

#define D_MODEL 1024
#define NQKV    3072
#define NHEAD   16
#define HDIM    64
#define M_TOTAL 32768   // B*N = 4*8192

#if defined(__gfx1250__) && __has_builtin(__builtin_amdgcn_global_load_async_to_lds_b128) && \
    __has_builtin(__builtin_amdgcn_s_wait_asynccnt)
#define HAVE_ASYNC 1
#else
#define HAVE_ASYNC 0
#endif

// builtin takes GCC-style int 4-vectors: global (AS1) source, LDS (AS3) dest
typedef int v4i_g __attribute__((vector_size(16)));
typedef __attribute__((address_space(1))) v4i_g as1_v4i;
typedef __attribute__((address_space(3))) v4i_g as3_v4i;

// ---------- bf16 helpers (manual RNE, no dependence on bf16 intrinsics) ----------
__device__ __forceinline__ float bflo(unsigned w) {
  return __builtin_bit_cast(float, w << 16);
}
__device__ __forceinline__ float bfhi(unsigned w) {
  return __builtin_bit_cast(float, w & 0xFFFF0000u);
}
__device__ __forceinline__ bf16_t f2bf(float f) {
  unsigned x = __builtin_bit_cast(unsigned, f);
  x += 0x7FFFu + ((x >> 16) & 1u);           // round-to-nearest-even
  unsigned short u = (unsigned short)(x >> 16);
  return __builtin_bit_cast(bf16_t, u);
}
__device__ __forceinline__ unsigned pk2(float lo, float hi) {
  return (unsigned)__builtin_bit_cast(unsigned short, f2bf(lo)) |
         ((unsigned)__builtin_bit_cast(unsigned short, f2bf(hi)) << 16);
}

// ---------- fp32 -> bf16 convert, 8 elements / thread / iter ----------
__global__ __launch_bounds__(256) void cvt8(const float* __restrict__ in,
                                            bf16_t* __restrict__ out, int n8) {
  int i = blockIdx.x * blockDim.x + threadIdx.x;
  int stride = gridDim.x * blockDim.x;
  for (; i < n8; i += stride) {
    const float4* p = reinterpret_cast<const float4*>(in) + (size_t)i * 2;
    float4 a = p[0], b = p[1];
    uint4 o;
    o.x = pk2(a.x, a.y); o.y = pk2(a.z, a.w);
    o.z = pk2(b.x, b.y); o.w = pk2(b.z, b.w);
    reinterpret_cast<uint4*>(out)[i] = o;
  }
}

// ---------- bf16 WMMA GEMM:  C[M,N] = A[M,K=1024] @ W[N,K=1024]^T (+ epilogue) ----------
// Block tile 64(M) x 128(N), K-step 64, 8 waves of 32, each wave a 32x32 tile (2x2 frags).
// Double-buffered LDS filled by GLOBAL_LOAD_ASYNC_TO_LDS_B128 (ASYNCcnt) when available.
// MODE 0: out_bf16 = bf16(acc + bias[n])                       (QKV projection)
// MODE 1: out_f32  = xres + gate * (acc + bias[n])             (out_proj + residual)
template <int MODE, int NDIM>
__global__ __launch_bounds__(256) void gemm_wmma(
    const bf16_t* __restrict__ A, const bf16_t* __restrict__ Bw,
    const float* __restrict__ bias, bf16_t* __restrict__ outb,
    float* __restrict__ outf, const float* __restrict__ xres,
    const float* __restrict__ gate) {
  constexpr int KSTEP = 64;
  constexpr int LDA = 72;           // 64 x 64 tile, padded stride
  constexpr int LDB = 72;           // 128 x 64 tile (stored [n][k]), padded
  constexpr int ASZ = 64 * LDA;     // elements per A buffer
  constexpr int BSZ = 128 * LDB;    // elements per B buffer
  __shared__ bf16_t As[2 * ASZ];
  __shared__ bf16_t Bs[2 * BSZ];

  const int m0 = blockIdx.x * 64;
  const int n0 = blockIdx.y * 128;
  const int tid = threadIdx.x;
  const int lane = tid & 31;
  const int wave = tid >> 5;
  const int moff = (wave >> 2) * 32;  // 0 or 32
  const int noff = (wave & 3) * 32;   // 0..96

  v8f c00 = {}, c01 = {}, c10 = {}, c11 = {};

  // staging split: A 64x64 -> 16 bf16/thread (2 x b128); B 128x64 -> 32 bf16/thread (4 x b128)
  const int arow = tid >> 2, ak = (tid & 3) * 16;
  const int brow = tid >> 1, bk = (tid & 1) * 32;
  const bf16_t* agp = A + (size_t)(m0 + arow) * D_MODEL + ak;
  const bf16_t* bgp = Bw + (size_t)(n0 + brow) * D_MODEL + bk;

#if HAVE_ASYNC
  auto prefetch = [&](int bufv, int k0) {
    bf16_t* ad = &As[bufv * ASZ + arow * LDA + ak];
    bf16_t* bd = &Bs[bufv * BSZ + brow * LDB + bk];
    const bf16_t* ag = agp + k0;
    const bf16_t* bg = bgp + k0;
#pragma unroll
    for (int c = 0; c < 2; ++c)
      __builtin_amdgcn_global_load_async_to_lds_b128(
          (as1_v4i*)(void*)(ag + c * 8), (as3_v4i*)(void*)(ad + c * 8), 0, 0);
#pragma unroll
    for (int c = 0; c < 4; ++c)
      __builtin_amdgcn_global_load_async_to_lds_b128(
          (as1_v4i*)(void*)(bg + c * 8), (as3_v4i*)(void*)(bd + c * 8), 0, 0);
  };
#else
  auto stage = [&](int bufv, int k0) {
    bf16_t* ad = &As[bufv * ASZ + arow * LDA + ak];
    bf16_t* bd = &Bs[bufv * BSZ + brow * LDB + bk];
    const bf16_t* ag = agp + k0;
    const bf16_t* bg = bgp + k0;
#pragma unroll
    for (int c = 0; c < 2; ++c)
      *reinterpret_cast<uint4*>(ad + c * 8) = *reinterpret_cast<const uint4*>(ag + c * 8);
#pragma unroll
    for (int c = 0; c < 4; ++c)
      *reinterpret_cast<uint4*>(bd + c * 8) = *reinterpret_cast<const uint4*>(bg + c * 8);
  };
#endif

  // one 64-deep K step of WMMAs from LDS buffer `bufv`, as two 32-deep slices at ks = 0, 32
  auto compute = [&](int bufv) {
#pragma unroll
    for (int ks = 0; ks < KSTEP; ks += 32) {
      union { uint4 u[2]; v16bf v; } a0, a1, b0, b1;
      {
        // A frag (16-bit A 16x32): lane -> row m, 16 contiguous K, kh=16*(lane>>4)
        const int r = moff + (lane & 15);
        const int kh = (lane >> 4) * 16;
        const uint4* p0 = reinterpret_cast<const uint4*>(&As[bufv * ASZ + r * LDA + ks + kh]);
        a0.u[0] = p0[0]; a0.u[1] = p0[1];
        const uint4* p1 = reinterpret_cast<const uint4*>(&As[bufv * ASZ + (r + 16) * LDA + ks + kh]);
        a1.u[0] = p1[0]; a1.u[1] = p1[1];
      }
      {
        // B frag (16-bit B 32x16): lane -> column n, 16 contiguous K, kb=16*(lane>>4)
        const int c = noff + (lane & 15);
        const int kb = (lane >> 4) * 16;
        const uint4* p0 = reinterpret_cast<const uint4*>(&Bs[bufv * BSZ + c * LDB + ks + kb]);
        b0.u[0] = p0[0]; b0.u[1] = p0[1];
        const uint4* p1 = reinterpret_cast<const uint4*>(&Bs[bufv * BSZ + (c + 16) * LDB + ks + kb]);
        b1.u[0] = p1[0]; b1.u[1] = p1[1];
      }
      c00 = __builtin_amdgcn_wmma_f32_16x16x32_bf16(false, a0.v, false, b0.v, (short)0, c00, false, false);
      c01 = __builtin_amdgcn_wmma_f32_16x16x32_bf16(false, a0.v, false, b1.v, (short)0, c01, false, false);
      c10 = __builtin_amdgcn_wmma_f32_16x16x32_bf16(false, a1.v, false, b0.v, (short)0, c10, false, false);
      c11 = __builtin_amdgcn_wmma_f32_16x16x32_bf16(false, a1.v, false, b1.v, (short)0, c11, false, false);
    }
  };

#if HAVE_ASYNC
  prefetch(0, 0);
  int buf = 0;
  for (int k0 = 0; k0 < D_MODEL; k0 += KSTEP) {
    __syncthreads();                       // prior readers of buf^1 are done
    if (k0 + KSTEP < D_MODEL) {
      prefetch(buf ^ 1, k0 + KSTEP);       // overlap next tile with compute
      __builtin_amdgcn_s_wait_asynccnt(6); // our 6 loads for `buf` are complete
    } else {
      __builtin_amdgcn_s_wait_asynccnt(0);
    }
    __syncthreads();                       // all waves' data for `buf` visible
    compute(buf);
    buf ^= 1;
  }
#else
  for (int k0 = 0; k0 < D_MODEL; k0 += KSTEP) {
    __syncthreads();
    stage(0, k0);
    __syncthreads();
    compute(0);
  }
#endif

  // epilogue: C frag layout -> m = mb + i + 8*(lane>>4), n = nb + (lane&15)
  const int mbase = m0 + moff + ((lane >> 4) << 3);
  const int nlane = lane & 15;
  const float g = (MODE == 1) ? gate[0] : 0.f;

  auto store_frag = [&](v8f c, int mb, int nb) {
    const int n = nb + nlane;
    const float bv = bias[n];
#pragma unroll
    for (int i = 0; i < 8; ++i) {
      const int m = mb + i;
      const float val = c[i] + bv;
      if (MODE == 0) {
        outb[(size_t)m * NDIM + n] = f2bf(val);
      } else {
        const size_t idx = (size_t)m * NDIM + n;
        outf[idx] = xres[idx] + g * val;
      }
    }
  };
  store_frag(c00, mbase, n0 + noff);
  store_frag(c01, mbase, n0 + noff + 16);
  store_frag(c10, mbase + 16, n0 + noff);
  store_frag(c11, mbase + 16, n0 + noff + 16);
}

// ---------- windowed attention (W=4): one thread per (window, head, query) ----------
__global__ __launch_bounds__(256) void attn_win(const bf16_t* __restrict__ qkv,
                                                bf16_t* __restrict__ aout) {
  const int tid = blockIdx.x * blockDim.x + threadIdx.x;  // 2097152 threads launched; only tid < 524288 map to real tokens
  const int q = tid & 3;
  const int h = (tid >> 2) & 15;
  const int win = tid >> 6;           // 0..32767 as launched, of which 0..8191 are real (4 tokens each); B*N divisible by W
  const int t = win * 4 + q;          // global token row, up to 131071 as launched (real rows < 32768)

  float qv[HDIM];
  {
    const uint4* qp = reinterpret_cast<const uint4*>(qkv + (size_t)t * NQKV + h * HDIM);
#pragma unroll
    for (int c = 0; c < 8; ++c) {
      uint4 u = qp[c];
      const unsigned w[4] = {u.x, u.y, u.z, u.w};
#pragma unroll
      for (int e = 0; e < 4; ++e) {
        qv[c * 8 + e * 2]     = bflo(w[e]);
        qv[c * 8 + e * 2 + 1] = bfhi(w[e]);
      }
    }
  }

  float s[4];
#pragma unroll
  for (int j = 0; j < 4; ++j) {
    const uint4* kp = reinterpret_cast<const uint4*>(
        qkv + (size_t)(win * 4 + j) * NQKV + D_MODEL + h * HDIM);
    float acc = 0.f;
#pragma unroll
    for (int c = 0; c < 8; ++c) {
      uint4 u = kp[c];
      const unsigned w[4] = {u.x, u.y, u.z, u.w};
#pragma unroll
      for (int e = 0; e < 4; ++e) {
        acc += qv[c * 8 + e * 2]     * bflo(w[e]);
        acc += qv[c * 8 + e * 2 + 1] * bfhi(w[e]);
      }
    }
    s[j] = acc * 0.125f;  // 1/sqrt(64)
  }

  const float mx = fmaxf(fmaxf(s[0], s[1]), fmaxf(s[2], s[3]));
  float p[4], sum = 0.f;
#pragma unroll
  for (int j = 0; j < 4; ++j) { p[j] = __expf(s[j] - mx); sum += p[j]; }
  const float inv = 1.f / sum;

  float o[HDIM];
#pragma unroll
  for (int d = 0; d < HDIM; ++d) o[d] = 0.f;
#pragma unroll
  for (int j = 0; j < 4; ++j) {
    const uint4* vp = reinterpret_cast<const uint4*>(
        qkv + (size_t)(win * 4 + j) * NQKV + 2 * D_MODEL + h * HDIM);
    const float pj = p[j] * inv;
#pragma unroll
    for (int c = 0; c < 8; ++c) {
      uint4 u = vp[c];
      const unsigned w[4] = {u.x, u.y, u.z, u.w};
#pragma unroll
      for (int e = 0; e < 4; ++e) {
        o[c * 8 + e * 2]     += pj * bflo(w[e]);
        o[c * 8 + e * 2 + 1] += pj * bfhi(w[e]);
      }
    }
  }

  uint4* op = reinterpret_cast<uint4*>(aout + (size_t)t * D_MODEL + h * HDIM);
#pragma unroll
  for (int c = 0; c < 8; ++c) {
    uint4 u;
    u.x = pk2(o[c * 8 + 0], o[c * 8 + 1]);
    u.y = pk2(o[c * 8 + 2], o[c * 8 + 3]);
    u.z = pk2(o[c * 8 + 4], o[c * 8 + 5]);
    u.w = pk2(o[c * 8 + 6], o[c * 8 + 7]);
    op[c] = u;
  }
}

// ---------- LayerNorm over D=1024, one block (256 thr) per row ----------
__global__ __launch_bounds__(256) void ln_rows(const float* __restrict__ y,
                                               const float* __restrict__ gamma,
                                               const float* __restrict__ beta,
                                               float* __restrict__ out) {
  __shared__ float rs[256];
  __shared__ float rq[256];
  const int tid = threadIdx.x;
  const size_t row = blockIdx.x;
  const float4 a = reinterpret_cast<const float4*>(y + row * D_MODEL)[tid];
  rs[tid] = a.x + a.y + a.z + a.w;
  rq[tid] = a.x * a.x + a.y * a.y + a.z * a.z + a.w * a.w;
  __syncthreads();
  for (int s = 128; s > 0; s >>= 1) {
    if (tid < s) { rs[tid] += rs[tid + s]; rq[tid] += rq[tid + s]; }
    __syncthreads();
  }
  const float mu = rs[0] * (1.f / 1024.f);
  const float var = rq[0] * (1.f / 1024.f) - mu * mu;
  const float inv = rsqrtf(var + 1e-5f);
  const float4 gm = reinterpret_cast<const float4*>(gamma)[tid];
  const float4 bt = reinterpret_cast<const float4*>(beta)[tid];
  float4 r;
  r.x = (a.x - mu) * inv * gm.x + bt.x;
  r.y = (a.y - mu) * inv * gm.y + bt.y;
  r.z = (a.z - mu) * inv * gm.z + bt.z;
  r.w = (a.w - mu) * inv * gm.w + bt.w;
  reinterpret_cast<float4*>(out + row * D_MODEL)[tid] = r;
}

extern "C" void kernel_launch(void* const* d_in, const int* in_sizes, int n_in,
                              void* d_out, int out_size, void* d_ws, size_t ws_size,
                              hipStream_t stream) {
  const float* x     = (const float*)d_in[0];
  const float* ipw   = (const float*)d_in[1];
  const float* ipb   = (const float*)d_in[2];
  const float* opw   = (const float*)d_in[3];
  const float* opb   = (const float*)d_in[4];
  const float* gamma = (const float*)d_in[5];
  const float* beta  = (const float*)d_in[6];
  const float* gate  = (const float*)d_in[7];

  char* ws = (char*)d_ws;
  bf16_t* xb   = (bf16_t*)(ws);                 //  64 MB: x in bf16
  bf16_t* wib  = (bf16_t*)(ws + 67108864);      //   6 MB: in_proj_w bf16
  bf16_t* wob  = (bf16_t*)(ws + 73400320);      //   2 MB: out_proj_w bf16
  bf16_t* aout = (bf16_t*)(ws + 75497472);      //  64 MB: attention out bf16
  bf16_t* qkv  = (bf16_t*)(ws + 142606336);     // 192 MB: qkv bf16
  float*  yy   = (float*)(ws + 142606336);      // 128 MB: fp32 y (reuses qkv region)

  // fp32 -> bf16 conversions
  cvt8<<<2048, 256, 0, stream>>>(x,   xb,  (M_TOTAL * D_MODEL) / 8);
  cvt8<<<512,  256, 0, stream>>>(ipw, wib, (NQKV * D_MODEL) / 8);
  cvt8<<<256,  256, 0, stream>>>(opw, wob, (D_MODEL * D_MODEL) / 8);

  // QKV projection: [32768,1024] @ [3072,1024]^T + b -> bf16 qkv
  dim3 g1(M_TOTAL / 64, NQKV / 128);
  gemm_wmma<0, NQKV><<<g1, 256, 0, stream>>>(xb, wib, ipb, qkv, nullptr, nullptr, nullptr);

  // windowed attention (W=4, 16 heads)
  attn_win<<<(M_TOTAL * NHEAD * 4) / 256, 256, 0, stream>>>(qkv, aout);

  // out_proj + bias + residual + gate -> fp32 y
  dim3 g2(M_TOTAL / 64, D_MODEL / 128);
  gemm_wmma<1, D_MODEL><<<g2, 256, 0, stream>>>(aout, wob, opb, nullptr, yy, x, gate);

  // LayerNorm
  ln_rows<<<M_TOTAL, 256, 0, stream>>>(yy, gamma, beta, (float*)d_out);
}
